// Block_12730283065401
// MI455X (gfx1250) — matrix operand order repaired and checked by reading, not compiled, not run on hardware
//
#include <hip/hip_runtime.h>

// ---------------------------------------------------------------------------
// Transformer block forward, MI455X (gfx1250).  bf16 WMMA, f32 accumulate.
// B=2, S=2048, D=1024, H=16 (hd=64), DFF=4096.
//
// CDNA5 paths used:
//   * v_wmma_f32_16x16x32_bf16 for all GEMMs + attention
//   * v_cvt_pk_bf16_f32 for all fp32->bf16 packing
//   * global_load_async_to_lds_b128 with double-buffered LDS ping-pong and
//     s_wait_asynccnt N pipelining for GEMM tile staging
//   * ds_* for the P-transpose in flash attention (wave-local, s_wait_dscnt)
// ---------------------------------------------------------------------------

#define B_SZ    2
#define S_LEN   2048
#define DMODEL  1024
#define NHEADS  16
#define HDIM    64
#define DFF     4096
#define MTOT    (B_SZ * S_LEN)      // 4096 rows

typedef __bf16  v16bf __attribute__((ext_vector_type(16)));
typedef float   v8f   __attribute__((ext_vector_type(8)));

union FragA { unsigned u[8]; v16bf v; };

// fp32 pair -> packed bf16 (single VALU op)
__device__ __forceinline__ unsigned cvtpk(float a, float b) {
  unsigned r;
  asm("v_cvt_pk_bf16_f32 %0, %1, %2" : "=v"(r) : "v"(a), "v"(b));
  return r;
}

// async 16B global -> LDS copy (per active lane), tracked by ASYNCcnt
__device__ __forceinline__ void async_cp16(unsigned lds_byte_addr, const void* gaddr) {
  asm volatile("global_load_async_to_lds_b128 %0, %1, off"
               :: "v"(lds_byte_addr), "v"(gaddr) : "memory");
}
template <int N>
__device__ __forceinline__ void wait_async() {
  asm volatile("s_wait_asynccnt %0" :: "i"(N) : "memory");
}
__device__ __forceinline__ unsigned lds_addr_of(const void* p) {
  return (unsigned)(unsigned long long)p;   // low 32 bits == LDS byte offset
}

// ---------------------------------------------------------------------------
// Pre-pack kernels: fp32 -> bf16-pair (uint) tensors.
// ---------------------------------------------------------------------------
__global__ void pack_contig(const float* __restrict__ src, unsigned* __restrict__ dst,
                            long npairs) {
  long i = (long)blockIdx.x * blockDim.x + threadIdx.x;
  if (i < npairs) {
    float2 f = *reinterpret_cast<const float2*>(src + 2 * i);
    dst[i] = cvtpk(f.x, f.y);
  }
}
// weights W[K,N] row-major -> Wpk[kp][n] = pack(W[2kp][n], W[2kp+1][n])
__global__ void pack_strided(const float* __restrict__ src, unsigned* __restrict__ dst,
                             int K, int N) {
  long i = (long)blockIdx.x * blockDim.x + threadIdx.x;
  long tot = (long)(K / 2) * N;
  if (i < tot) {
    int kp = (int)(i / N), n = (int)(i % N);
    dst[i] = cvtpk(src[(size_t)(2 * kp) * N + n], src[(size_t)(2 * kp + 1) * N + n]);
  }
}

// ---------------------------------------------------------------------------
// Tiled GEMM on pre-packed bf16 pairs, double-buffered async-to-LDS staging.
//   Apk: M x K/2 uints (pairs along K);  Wpk: K/2 x N uints (pair-major rows)
// WG tile 128x64, BK=32, 256 thr = 8 waves, wave = 16x64 strip (4 wmma accs).
// PK output mode: 0=none, 1=bf16 pairs along N, 2=bf16 pairs along M.
// ---------------------------------------------------------------------------
enum { EPI_BIAS = 0, EPI_GELU = 1, EPI_RES = 2 };

template <int EPI, bool WF32, int PK>
__global__ __launch_bounds__(256) void gemm_bf16_wmma(
    const unsigned* __restrict__ Apk, const unsigned* __restrict__ Wpk,
    const float* __restrict__ bias, const float* __restrict__ Res,
    float* __restrict__ Outf, unsigned* __restrict__ Outpk,
    int M, int N, int K)
{
  __shared__ __align__(16) unsigned As[2][128 * 16]; // [row][kp^sw], 16B-chunk swizzle
  __shared__ __align__(16) unsigned Bs[2][16 * 64];  // [kp][n ^ (g<<4)] chunk swizzle

  const int tid  = threadIdx.x;
  const int m0   = blockIdx.y * 128;
  const int n0   = blockIdx.x * 64;
  const int lane = tid & 31;
  const int wv   = tid >> 5;
  const int g    = lane >> 4;
  const int c16  = lane & 15;
  const int Kp   = K >> 1;

  // per-stage issue: 3 async_cp16 per lane (2 for A, 1 for B)
  auto issue_stage = [&](int k0, int buf) {
    const int kp0 = k0 >> 1;
    const unsigned asb = lds_addr_of(&As[buf][0]);
    const unsigned bsb = lds_addr_of(&Bs[buf][0]);
    #pragma unroll
    for (int i = 0; i < 2; ++i) {
      int c   = tid + 256 * i;          // 0..511 chunks of 16B (A: 128x16 uints)
      int row = c >> 2;
      int j   = c & 3;
      int sw  = ((row >> 2) & 3) << 2;
      const void* g4 = Apk + (size_t)(m0 + row) * Kp + kp0 + 4 * j;
      async_cp16(asb + (unsigned)(row * 16 + ((4 * j) ^ sw)) * 4u, g4);
    }
    {
      int kp = tid >> 4;                // 256 chunks (B: 16x64 uints)
      int j  = tid & 15;
      const void* g4 = Wpk + (size_t)(kp0 + kp) * N + n0 + 4 * j;
      async_cp16(bsb + (unsigned)(kp * 64 + 4 * (j ^ ((kp >> 3) << 2))) * 4u, g4);
    }
  };

  v8f acc[4];
  v8f zf = {};
  #pragma unroll
  for (int t = 0; t < 4; ++t) acc[t] = zf;

  const int NT = K >> 5;          // K/32 stages
  issue_stage(0, 0);

  for (int st = 0; st < NT; ++st) {
    const int cur = st & 1;
    if (st + 1 < NT) {            // prefetch next stage into the other buffer
      issue_stage(32 * (st + 1), cur ^ 1);
      wait_async<3>();            // leave only the 3 just-issued outstanding
    } else {
      wait_async<0>();
    }
    __syncthreads();

    // ---- A fragment (16x32 bf16, contiguous k = 16g + s, same order as B)
    FragA af;
    const int rl = 16 * wv + c16;
    const int sw = ((rl >> 2) & 3) << 2;
    #pragma unroll
    for (int v = 0; v < 8; ++v) {
      int kp = 8 * g + v;
      af.u[v] = As[cur][rl * 16 + (kp ^ sw)];
    }
    // ---- 4 x (B fragment + WMMA)
    #pragma unroll
    for (int t = 0; t < 4; ++t) {
      FragA bf;
      #pragma unroll
      for (int v = 0; v < 8; ++v) {
        int kp = 8 * g + v;
        bf.u[v] = Bs[cur][kp * 64 + (((16 * t + c16) ^ (g << 4)))];
      }
      acc[t] = __builtin_amdgcn_wmma_f32_16x16x32_bf16(
          false, af.v, false, bf.v, (short)0, acc[t], false, false);
    }
    __syncthreads();              // frees buf[cur] for the issue in st+1
  }

  // ---- epilogue (C layout: m = v + 8g, n = 16t + c16)
  #pragma unroll
  for (int t = 0; t < 4; ++t) {
    float vals[8];
    #pragma unroll
    for (int v = 0; v < 8; ++v) {
      int nc = n0 + 16 * t + c16;
      float val = acc[t][v] + bias[nc];
      if (EPI == EPI_GELU) val = 0.5f * val * (1.0f + erff(val * 0.70710678118654752f));
      if (EPI == EPI_RES)  val += Res[(size_t)(m0 + 16 * wv + v + 8 * g) * N + nc];
      vals[v] = val;
    }
    #pragma unroll
    for (int v = 0; v < 8; ++v) {
      int mr = m0 + 16 * wv + v + 8 * g;
      int nc = n0 + 16 * t + c16;
      if (WF32) Outf[(size_t)mr * N + nc] = vals[v];
      if (PK == 1) {              // pairs along N (cross-lane)
        float pv = __shfl_xor(vals[v], 1, 32);
        if ((c16 & 1) == 0)
          Outpk[(size_t)mr * (N >> 1) + (nc >> 1)] = cvtpk(vals[v], pv);
      }
      if (PK == 2 && (v & 1) == 0) {  // pairs along M (same lane, adjacent accs)
        Outpk[(size_t)(mr >> 1) * N + nc] = cvtpk(vals[v], vals[v + 1]);
      }
    }
  }
}

// ---------------------------------------------------------------------------
// Flash attention (causal).  Wave = 16-query tile of one (b,h); streams 32-key
// blocks.  Q/K pre-packed pairs along D;  V pre-packed pairs along S.
// ---------------------------------------------------------------------------
__global__ __launch_bounds__(256) void flash_attn(
    const unsigned* __restrict__ Qpk, const unsigned* __restrict__ Kpk,
    const unsigned* __restrict__ Vpk, unsigned* __restrict__ Opk)
{
  __shared__ unsigned Pls[8 * 16 * 16];   // per-wave 16 rows x 16 key-pairs

  const int lane = threadIdx.x & 31;
  const int wv   = threadIdx.x >> 5;
  const int g    = lane >> 4;
  const int c16  = lane & 15;
  const int b    = blockIdx.y / NHEADS;
  const int h    = blockIdx.y % NHEADS;
  const int q0   = (blockIdx.x * 8 + wv) * 16;
  const int Dp   = DMODEL >> 1;

  const unsigned* Qh = Qpk + (size_t)b * S_LEN * Dp + h * (HDIM / 2);
  const unsigned* Kh = Kpk + (size_t)b * S_LEN * Dp + h * (HDIM / 2);
  const unsigned* Vh = Vpk + (size_t)b * (S_LEN / 2) * DMODEL + h * HDIM;
  unsigned*       Oh = Opk + (size_t)b * S_LEN * Dp + h * (HDIM / 2);

  // Q fragments: 16x64 as two 16x32 A fragments (direct packed loads)
  FragA qf[2];
  {
    const unsigned* qrow = Qh + (size_t)(q0 + c16) * Dp;
    #pragma unroll
    for (int kc = 0; kc < 2; ++kc)
      #pragma unroll
      for (int v = 0; v < 8; ++v) {
        int kp = 16 * kc + 8 * g + v;
        qf[kc].u[v] = qrow[kp];
      }
  }

  float mi[8], li[8];
  #pragma unroll
  for (int v = 0; v < 8; ++v) { mi[v] = -__builtin_inff(); li[v] = 0.0f; }
  v8f o[4];
  v8f zf = {};
  #pragma unroll
  for (int t = 0; t < 4; ++t) o[t] = zf;

  for (int kb = 0; kb < q0 + 16; kb += 32) {
    // ---- S[16x32] = Q @ K^T  (B[kk][n] = K[key][kk], packed loads)
    v8f s[2]; s[0] = zf; s[1] = zf;
    #pragma unroll
    for (int nt = 0; nt < 2; ++nt) {
      const unsigned* krow = Kh + (size_t)(kb + 16 * nt + c16) * Dp;
      #pragma unroll
      for (int kc = 0; kc < 2; ++kc) {
        FragA kf;
        #pragma unroll
        for (int v = 0; v < 8; ++v) kf.u[v] = krow[16 * kc + 8 * g + v];
        s[nt] = __builtin_amdgcn_wmma_f32_16x16x32_bf16(
            false, qf[kc].v, false, kf.v, (short)0, s[nt], false, false);
      }
    }

    // ---- scale, causal mask, online softmax (rows at m = v + 8g)
    float mnew[8], fac[8];
    #pragma unroll
    for (int v = 0; v < 8; ++v) {
      const int qrow = q0 + v + 8 * g;
      #pragma unroll
      for (int nt = 0; nt < 2; ++nt) {
        int key = kb + 16 * nt + c16;
        float sv = s[nt][v] * 0.125f;               // 1/sqrt(64)
        s[nt][v] = (key > qrow) ? -__builtin_inff() : sv;
      }
      float t0 = fmaxf(s[0][v], s[1][v]);
      t0 = fmaxf(t0, __shfl_xor(t0, 1, 32));
      t0 = fmaxf(t0, __shfl_xor(t0, 2, 32));
      t0 = fmaxf(t0, __shfl_xor(t0, 4, 32));
      t0 = fmaxf(t0, __shfl_xor(t0, 8, 32));
      mnew[v] = fmaxf(mi[v], t0);
      fac[v]  = __expf(mi[v] - mnew[v]);
      mi[v]   = mnew[v];
    }
    #pragma unroll
    for (int v = 0; v < 8; ++v) {
      float p0 = __expf(s[0][v] - mnew[v]);
      float p1 = __expf(s[1][v] - mnew[v]);
      s[0][v] = p0; s[1][v] = p1;
      float r = p0 + p1;
      r += __shfl_xor(r, 1, 32);
      r += __shfl_xor(r, 2, 32);
      r += __shfl_xor(r, 4, 32);
      r += __shfl_xor(r, 8, 32);
      li[v] = li[v] * fac[v] + r;
      #pragma unroll
      for (int t = 0; t < 4; ++t) o[t][v] *= fac[v];
    }

    // ---- transpose P (C layout) -> A layout via wave-local LDS (packed pairs)
    unsigned* pw = &Pls[wv * 256];
    #pragma unroll
    for (int nt = 0; nt < 2; ++nt)
      #pragma unroll
      for (int v = 0; v < 8; ++v) {
        float pv = s[nt][v];
        float pp = __shfl_xor(pv, 1, 32);
        if ((c16 & 1) == 0)
          pw[(v + 8 * g) * 16 + 8 * nt + (c16 >> 1)] = cvtpk(pv, pp);
      }
    asm volatile("s_wait_dscnt 0" ::: "memory");
    FragA pf;
    #pragma unroll
    for (int v = 0; v < 8; ++v) {
      int kp = 8 * g + v;
      pf.u[v] = pw[c16 * 16 + kp];
    }

    // ---- O += P @ V  (V packed along S: one uint load per VGPR)
    #pragma unroll
    for (int t = 0; t < 4; ++t) {
      FragA vf;
      #pragma unroll
      for (int v = 0; v < 8; ++v) {
        vf.u[v] = Vh[(size_t)((kb >> 1) + 8 * g + v) * DMODEL + 16 * t + c16];
      }
      o[t] = __builtin_amdgcn_wmma_f32_16x16x32_bf16(
          false, pf.v, false, vf.v, (short)0, o[t], false, false);
    }
  }

  // ---- finalize: /rowsum, emit packed bf16 context (pairs along D)
  #pragma unroll
  for (int v = 0; v < 8; ++v) {
    float inv = 1.0f / li[v];
    int row = q0 + v + 8 * g;
    #pragma unroll
    for (int t = 0; t < 4; ++t) {
      float val = o[t][v] * inv;
      float pv  = __shfl_xor(val, 1, 32);
      if ((c16 & 1) == 0)
        Oh[(size_t)row * Dp + (16 * t + c16) / 2] = cvtpk(val, pv);
    }
  }
}

// ---------------------------------------------------------------------------
// Orchestration.
// ---------------------------------------------------------------------------
extern "C" void kernel_launch(void* const* d_in, const int* in_sizes, int n_in,
                              void* d_out, int out_size, void* d_ws, size_t ws_size,
                              hipStream_t stream) {
  (void)in_sizes; (void)n_in; (void)out_size; (void)ws_size;
  const float* x  = (const float*)d_in[0];
  const float* Wq = (const float*)d_in[1];
  const float* bq = (const float*)d_in[2];
  const float* Wk = (const float*)d_in[3];
  const float* bk = (const float*)d_in[4];
  const float* Wv = (const float*)d_in[5];
  const float* bv = (const float*)d_in[6];
  const float* Wo = (const float*)d_in[7];
  const float* bo = (const float*)d_in[8];
  const float* Wp = (const float*)d_in[9];
  const float* bp = (const float*)d_in[10];
  const float* W1 = (const float*)d_in[11];
  const float* b1 = (const float*)d_in[12];
  const float* W2 = (const float*)d_in[13];
  const float* b2 = (const float*)d_in[14];
  float* out = (float*)d_out;

  // ---- carve workspace (4-byte units)
  unsigned* ws = (unsigned*)d_ws;
  const size_t PD  = (size_t)MTOT * (DMODEL / 2);   // packed M x D
  const size_t PW  = (size_t)(DMODEL / 2) * DMODEL; // packed D-weights
  unsigned* xpk   = ws;                 ws += PD;
  unsigned* wqpk  = ws;                 ws += PW;
  unsigned* wkpk  = ws;                 ws += PW;
  unsigned* wvpk  = ws;                 ws += PW;
  unsigned* wopk  = ws;                 ws += PW;
  unsigned* wppk  = ws;                 ws += PW;
  unsigned* w1pk  = ws;                 ws += (size_t)(DMODEL / 2) * DFF;
  unsigned* w2pk  = ws;                 ws += (size_t)(DFF / 2) * DMODEL;
  unsigned* qpk   = ws;                 ws += PD;
  unsigned* kpk   = ws;                 ws += PD;
  unsigned* vspk  = ws;                 ws += PD;   // V packed along S: (M/2) x D
  unsigned* attnpk= ws;                 ws += PD;
  unsigned* tmppk = ws;                 ws += PD;
  float*    hresf = (float*)ws;         ws += (size_t)MTOT * DMODEL;
  unsigned* hrespk= ws;                 ws += PD;
  unsigned* ff1pk = ws;                 /* M x DFF/2 */

  dim3 blk(256);
  dim3 gD(DMODEL / 64, MTOT / 128);       // (16, 32)
  dim3 gF1(DFF / 64, MTOT / 128);         // (64, 32)
  dim3 gA(S_LEN / 16 / 8, B_SZ * NHEADS); // (16, 32)

  // ---- pre-pack x and all weights to bf16 pairs
  {
    long np = (long)MTOT * DMODEL / 2;
    pack_contig<<<(np + 255) / 256, blk, 0, stream>>>(x, xpk, np);
    long nw = (long)(DMODEL / 2) * DMODEL;
    pack_strided<<<(nw + 255) / 256, blk, 0, stream>>>(Wq, wqpk, DMODEL, DMODEL);
    pack_strided<<<(nw + 255) / 256, blk, 0, stream>>>(Wk, wkpk, DMODEL, DMODEL);
    pack_strided<<<(nw + 255) / 256, blk, 0, stream>>>(Wv, wvpk, DMODEL, DMODEL);
    pack_strided<<<(nw + 255) / 256, blk, 0, stream>>>(Wo, wopk, DMODEL, DMODEL);
    pack_strided<<<(nw + 255) / 256, blk, 0, stream>>>(Wp, wppk, DMODEL, DMODEL);
    long n1 = (long)(DMODEL / 2) * DFF;
    pack_strided<<<(n1 + 255) / 256, blk, 0, stream>>>(W1, w1pk, DMODEL, DFF);
    long n2 = (long)(DFF / 2) * DMODEL;
    pack_strided<<<(n2 + 255) / 256, blk, 0, stream>>>(W2, w2pk, DFF, DMODEL);
  }

  // ---- q,k packed along D; v packed along S (direct from GEMM epilogue)
  gemm_bf16_wmma<EPI_BIAS, false, 1><<<gD, blk, 0, stream>>>(
      xpk, wqpk, bq, nullptr, nullptr, qpk, MTOT, DMODEL, DMODEL);
  gemm_bf16_wmma<EPI_BIAS, false, 1><<<gD, blk, 0, stream>>>(
      xpk, wkpk, bk, nullptr, nullptr, kpk, MTOT, DMODEL, DMODEL);
  gemm_bf16_wmma<EPI_BIAS, false, 2><<<gD, blk, 0, stream>>>(
      xpk, wvpk, bv, nullptr, nullptr, vspk, MTOT, DMODEL, DMODEL);
  // ---- attention -> packed context
  flash_attn<<<gA, blk, 0, stream>>>(qpk, kpk, vspk, attnpk);
  // ---- Wo -> packed;  Wp(+x residual) -> fp32 + packed
  gemm_bf16_wmma<EPI_BIAS, false, 1><<<gD, blk, 0, stream>>>(
      attnpk, wopk, bo, nullptr, nullptr, tmppk, MTOT, DMODEL, DMODEL);
  gemm_bf16_wmma<EPI_RES,  true,  1><<<gD, blk, 0, stream>>>(
      tmppk, wppk, bp, x, hresf, hrespk, MTOT, DMODEL, DMODEL);
  // ---- FFN: GELU -> packed;  final (+hres residual) -> d_out fp32
  gemm_bf16_wmma<EPI_GELU, false, 1><<<gF1, blk, 0, stream>>>(
      hrespk, w1pk, b1, nullptr, nullptr, ff1pk, MTOT, DFF, DMODEL);
  gemm_bf16_wmma<EPI_RES,  true,  0><<<gD, blk, 0, stream>>>(
      ff1pk, w2pk, b2, hresf, out, nullptr, MTOT, DMODEL, DFF);
}
